// AAEncoder_52767968199106
// MI455X (gfx1250) — matrix operand order repaired and checked by reading, not compiled, not run on hardware
//
#include <hip/hip_runtime.h>

// ---------------------------------------------------------------------------
// AAEncoder fused kernel for gfx1250 (MI455X), wave32 + WMMA f16->f32.
//   K0: per-agent precompute (R, pos_t, padf, center, q*1/sqrt(8), e1)
//   K1: f32 -> f16 weight conversion (ne0_W2, na_W, Wk, Wv)
//   K2: fused per-row pipeline: e0 -> nbr -> k/v (WMMA) -> flash attention
//       -> Wo/gate/LN/MLP epilogue
// ---------------------------------------------------------------------------

#define NAG 1024
#define TDIM 50
#define DD 64
#define LNEPS 1e-5f

typedef __attribute__((ext_vector_type(16))) _Float16 v16h;
typedef __attribute__((ext_vector_type(8)))  _Float16 v8h;
typedef __attribute__((ext_vector_type(8)))  float    v8f;

// ======================== K0: per-agent precompute ==========================

struct K0Args {
    const float* positions;            // (N,T,2)
    const unsigned char* pmask;        // (N,T) bool
    const int* tptr;
    const float *ceW1, *ceb1, *ceg1, *cebe1;
    const float *ceW2, *ceb2, *ceg2, *cebe2;
    const float *ceW3, *ceb3, *ceg3, *cebe3;
    const float *n1g, *n1b, *Wq;
    const float *ne1W1, *ne1b1, *ne1g1, *ne1be1, *ne1W2, *ne1b2;
    float *Rws, *posws, *padws, *ctrws, *qws, *e1ws;
};

__device__ __forceinline__ float ln64k(float x, float g, float b, float* sh, int f) {
    __syncthreads();
    sh[f] = x;
    __syncthreads();
    float m = 0.f;
    for (int k = 0; k < 64; ++k) m += sh[k];
    m *= (1.f / 64.f);
    float v = 0.f;
    for (int k = 0; k < 64; ++k) { float d = sh[k] - m; v += d * d; }
    v *= (1.f / 64.f);
    return (x - m) * rsqrtf(v + LNEPS) * g + b;
}

__device__ __forceinline__ float dot64k(float x, const float* Wrow, float bias, float* sh, int f) {
    __syncthreads();
    sh[f] = x;
    __syncthreads();
    float s = bias;
    for (int k = 0; k < 64; ++k) s += sh[k] * Wrow[k];
    return s;
}

__global__ __launch_bounds__(64) void aae_precompute(K0Args A) {
    __shared__ float sh[64];
    const int i = blockIdx.x;
    const int f = threadIdx.x;
    const int tt = *A.tptr;

    const float p0x = A.positions[(i * TDIM + tt) * 2 + 0];
    const float p0y = A.positions[(i * TDIM + tt) * 2 + 1];
    const float pmx = A.positions[(i * TDIM + tt - 1) * 2 + 0];
    const float pmy = A.positions[(i * TDIM + tt - 1) * 2 + 1];
    const float dx = p0x - pmx, dy = p0y - pmy;
    const float ang = atan2f(dy, dx);
    const float c = cosf(ang), s = sinf(ang);
    // rotated dpos: dpos @ R[i]
    const float dr0 = dx * c + dy * s;
    const float dr1 = -dx * s + dy * c;

    if (f == 0) {
        A.Rws[4 * i + 0] = c;  A.Rws[4 * i + 1] = -s;
        A.Rws[4 * i + 2] = s;  A.Rws[4 * i + 3] = c;
        A.posws[2 * i + 0] = p0x;
        A.posws[2 * i + 1] = p0y;
        A.padws[i] = A.pmask[i * TDIM + tt] ? 1.0f : 0.0f;
    }

    // ---- center encoder ----
    float x = dr0 * A.ceW1[2 * f] + dr1 * A.ceW1[2 * f + 1] + A.ceb1[f];
    x = fmaxf(ln64k(x, A.ceg1[f], A.cebe1[f], sh, f), 0.f);
    x = dot64k(x, A.ceW2 + f * 64, A.ceb2[f], sh, f);
    x = fmaxf(ln64k(x, A.ceg2[f], A.cebe2[f], sh, f), 0.f);
    x = dot64k(x, A.ceW3 + f * 64, A.ceb3[f], sh, f);
    x = ln64k(x, A.ceg3[f], A.cebe3[f], sh, f);
    x = ln64k(x, A.n1g[f], A.n1b[f], sh, f);          // center (post n1)
    A.ctrws[i * 64 + f] = x;
    float qv = dot64k(x, A.Wq + f * 64, 0.0f, sh, f) * 0.35355339059f; // 1/sqrt(DH)
    A.qws[i * 64 + f] = qv;

    // ---- e1 (depends only on this agent's raw dpos) ----
    float xa = dx * A.ne1W1[2 * f] + dy * A.ne1W1[2 * f + 1] + A.ne1b1[f];
    xa = fmaxf(ln64k(xa, A.ne1g1[f], A.ne1be1[f], sh, f), 0.f);
    float e1v = dot64k(xa, A.ne1W2 + f * 64, A.ne1b2[f], sh, f);
    A.e1ws[i * 64 + f] = e1v;
}

// ======================== K1: f16 weight conversion =========================

__global__ __launch_bounds__(256) void aae_cvt_f16(const float* a, const float* b,
                                                   const float* c, const float* d,
                                                   _Float16* o) {
    int idx = blockIdx.x * blockDim.x + threadIdx.x;
    if (idx < 4096) {
        o[idx]         = (_Float16)a[idx];
        o[4096 + idx]  = (_Float16)b[idx];
        o[8192 + idx]  = (_Float16)c[idx];
        o[12288 + idx] = (_Float16)d[idx];
    }
}

// ======================== K2: fused main kernel =============================

struct MainArgs {
    const float *pos, *padf, *R, *center, *q, *e1;
    const _Float16 *wNe0, *wNa, *wWk, *wWv;   // f16, row-major 64x64 each
    const float *ne0W1, *ne0b1, *ne0g1, *ne0be1, *ne0b2;
    const float *nag1, *nabe1, *nab, *nag2, *nabe2;
    const float *Wo, *selfW, *selfB, *ihW, *ihB, *hhW, *hhB;
    const float *n2g, *n2b, *mW1, *mB1, *mW2, *mB2;
    float* out;
};

// LN over the 64 features of a row held by lane pair (l, l^16), relu opt,
// pack into two f16 A-fragments (kstep 0 / kstep 1 of 16x16x32 WMMA).
__device__ __forceinline__ void ln_frag(const float* xv0, const float* xv1,
                                        const int fidx[2][16],
                                        const float* g, const float* b,
                                        bool do_relu, v16h& a0, v16h& a1) {
    float sum = 0.f, sq = 0.f;
#pragma unroll
    for (int e = 0; e < 16; ++e) {
        sum += xv0[e] + xv1[e];
        sq  += xv0[e] * xv0[e] + xv1[e] * xv1[e];
    }
    sum += __shfl_xor(sum, 16, 32);
    sq  += __shfl_xor(sq, 16, 32);
    const float m   = sum * (1.f / 64.f);
    const float var = fmaxf(sq * (1.f / 64.f) - m * m, 0.f);
    const float r   = rsqrtf(var + LNEPS);
#pragma unroll
    for (int e = 0; e < 16; ++e) {
        float y0 = (xv0[e] - m) * r * g[fidx[0][e]] + b[fidx[0][e]];
        float y1 = (xv1[e] - m) * r * g[fidx[1][e]] + b[fidx[1][e]];
        if (do_relu) { y0 = fmaxf(y0, 0.f); y1 = fmaxf(y1, 0.f); }
        a0[e] = (_Float16)y0;
        a1[e] = (_Float16)y1;
    }
}

// D(16x64) = A(16x64) x W^T via 4 N-tiles x 2 K-steps of wmma_f32_16x16x32_f16.
// B-fragment: lane l -> row nt*16+(l&15) of W, cols kstep*32+(l>>4)*8 .. +7 and +16 .. +23.
__device__ __forceinline__ void gemm64(v16h a0, v16h a1, const _Float16* W,
                                       int l15, int lhi, v8f* acc) {
#pragma unroll
    for (int nt = 0; nt < 4; ++nt) {
        {
            const _Float16* p = W + (nt * 16 + l15) * 64 + (lhi * 8);
            v8h lo = *(const v8h*)p;
            v8h hi = *(const v8h*)(p + 16);
            v16h bb;
#pragma unroll
            for (int e = 0; e < 8; ++e) { bb[e] = lo[e]; bb[8 + e] = hi[e]; }
            acc[nt] = __builtin_amdgcn_wmma_f32_16x16x32_f16(
                false, a0, false, bb, (short)0, acc[nt], false, false);
        }
        {
            const _Float16* p = W + (nt * 16 + l15) * 64 + (32 + lhi * 8);
            v8h lo = *(const v8h*)p;
            v8h hi = *(const v8h*)(p + 16);
            v16h bb;
#pragma unroll
            for (int e = 0; e < 8; ++e) { bb[e] = lo[e]; bb[8 + e] = hi[e]; }
            acc[nt] = __builtin_amdgcn_wmma_f32_16x16x32_f16(
                false, a1, false, bb, (short)0, acc[nt], false, false);
        }
    }
}

__global__ __launch_bounds__(256) void aae_main(MainArgs A) {
    // region: per-wave 16x68 f32 relayout tiles, reused as f16 k/v (128x68 each),
    // reused again as epilogue scratch.  34816 bytes.
    __shared__ float region[8704];
    __shared__ float cbuf[704];     // 11 x 64 constants
    __shared__ float maskf[128];
    __shared__ float mhabuf[64];

    const int tid  = threadIdx.x;
    const int lane = tid & 31;
    const int wave = tid >> 5;
    const int l15  = lane & 15;
    const int lhi  = lane >> 4;
    const int i    = blockIdx.x;

    if (tid < 64) {
        cbuf[tid]       = A.ne0W1[2 * tid];
        cbuf[64 + tid]  = A.ne0W1[2 * tid + 1];
        cbuf[128 + tid] = A.ne0b1[tid];
        cbuf[192 + tid] = A.ne0g1[tid];
        cbuf[256 + tid] = A.ne0be1[tid];
        cbuf[320 + tid] = A.ne0b2[tid];
        cbuf[384 + tid] = A.nag1[tid];
        cbuf[448 + tid] = A.nabe1[tid];
        cbuf[512 + tid] = A.nab[tid];
        cbuf[576 + tid] = A.nag2[tid];
        cbuf[640 + tid] = A.nabe2[tid];
    }
    __syncthreads();

    const float pix = A.pos[2 * i], piy = A.pos[2 * i + 1];
    const float ci = A.R[4 * i + 0];   // cos
    const float si = A.R[4 * i + 2];   // sin

    float qh[8];
#pragma unroll
    for (int d = 0; d < 8; ++d) qh[d] = A.q[i * 64 + wave * 8 + d];

    int fidx[2][16];
#pragma unroll
    for (int s = 0; s < 2; ++s)
#pragma unroll
        for (int e = 0; e < 16; ++e)
            fidx[s][e] = s * 32 + ((e >> 3) << 4) + (lhi << 3) + (e & 7);

    float*     xt = region + wave * (16 * 68);
    _Float16*  kb = (_Float16*)region;           // 128 x 68 halves
    _Float16*  vb = kb + 128 * 68;               // 128 x 68 halves

    // flash-attention state for head `wave`, distributed over 32 lanes
    float fm = -1e30f, fl = 0.f, facc[8];
#pragma unroll
    for (int d = 0; d < 8; ++d) facc[d] = 0.f;

    for (int ch = 0; ch < 8; ++ch) {
        const int jg = ch * 128 + wave * 16 + l15;
        const float rjx = A.pos[2 * jg + 0] - pix;
        const float rjy = A.pos[2 * jg + 1] - piy;
        const float rr0 =  rjx * ci + rjy * si;
        const float rr1 = -rjx * si + rjy * ci;
        if (lane < 16) {
            const bool ok = ((rjx * rjx + rjy * rjy) <= 2500.0f &&
                             A.padf[jg] == 0.0f) || (jg == i);
            maskf[wave * 16 + l15] = ok ? 0.0f : -1e30f;
        }

        // ---- stage 1: linear(2->64) + LN + relu, in A-layout ----
        float xv0[16], xv1[16];
#pragma unroll
        for (int e = 0; e < 16; ++e) {
            const int f0 = fidx[0][e], f1 = fidx[1][e];
            xv0[e] = rr0 * cbuf[f0] + rr1 * cbuf[64 + f0] + cbuf[128 + f0];
            xv1[e] = rr0 * cbuf[f1] + rr1 * cbuf[64 + f1] + cbuf[128 + f1];
        }
        v16h a0, a1;
        ln_frag(xv0, xv1, fidx, cbuf + 192, cbuf + 256, true, a0, a1);

        v8f acc[4];
#pragma unroll
        for (int nt = 0; nt < 4; ++nt)
#pragma unroll
            for (int r = 0; r < 8; ++r) acc[nt][r] = 0.f;
        gemm64(a0, a1, A.wNe0, l15, lhi, acc);       // e0 (pre-bias)

        // ---- C-layout -> LDS, reload in A-layout, + b2 + e1, LN, relu ----
#pragma unroll
        for (int nt = 0; nt < 4; ++nt)
#pragma unroll
            for (int r = 0; r < 8; ++r)
                xt[(lhi * 8 + r) * 68 + nt * 16 + l15] = acc[nt][r];
        __syncthreads();
#pragma unroll
        for (int e = 0; e < 16; ++e) {
            const int f0 = fidx[0][e], f1 = fidx[1][e];
            xv0[e] = xt[l15 * 68 + f0] + cbuf[320 + f0] + A.e1[jg * 64 + f0];
            xv1[e] = xt[l15 * 68 + f1] + cbuf[320 + f1] + A.e1[jg * 64 + f1];
        }
        ln_frag(xv0, xv1, fidx, cbuf + 384, cbuf + 448, true, a0, a1);
        __syncthreads();

#pragma unroll
        for (int nt = 0; nt < 4; ++nt)
#pragma unroll
            for (int r = 0; r < 8; ++r) acc[nt][r] = 0.f;
        gemm64(a0, a1, A.wNa, l15, lhi, acc);        // nbr pre-LN

#pragma unroll
        for (int nt = 0; nt < 4; ++nt)
#pragma unroll
            for (int r = 0; r < 8; ++r)
                xt[(lhi * 8 + r) * 68 + nt * 16 + l15] = acc[nt][r];
        __syncthreads();
#pragma unroll
        for (int e = 0; e < 16; ++e) {
            const int f0 = fidx[0][e], f1 = fidx[1][e];
            xv0[e] = xt[l15 * 68 + f0] + cbuf[512 + f0];
            xv1[e] = xt[l15 * 68 + f1] + cbuf[512 + f1];
        }
        ln_frag(xv0, xv1, fidx, cbuf + 576, cbuf + 640, false, a0, a1); // nbr
        __syncthreads();   // all xt reads done before k/v overwrite region

        // ---- k = nbr@Wk^T, v = nbr@Wv^T ----
        v8f kacc[4], vacc[4];
#pragma unroll
        for (int nt = 0; nt < 4; ++nt)
#pragma unroll
            for (int r = 0; r < 8; ++r) { kacc[nt][r] = 0.f; vacc[nt][r] = 0.f; }
        gemm64(a0, a1, A.wWk, l15, lhi, kacc);
        gemm64(a0, a1, A.wWv, l15, lhi, vacc);
#pragma unroll
        for (int nt = 0; nt < 4; ++nt)
#pragma unroll
            for (int r = 0; r < 8; ++r) {
                const int row = wave * 16 + lhi * 8 + r;
                const int col = nt * 16 + l15;
                kb[row * 68 + col] = (_Float16)kacc[nt][r];
                vb[row * 68 + col] = (_Float16)vacc[nt][r];
            }
        __syncthreads();

        // ---- attention update: wave == head ----
        const int hb = wave * 8;
#pragma unroll
        for (int it2 = 0; it2 < 4; ++it2) {
            const int rr = it2 * 32 + lane;
            float lg = 0.f;
#pragma unroll
            for (int d = 0; d < 8; ++d)
                lg += (float)kb[rr * 68 + hb + d] * qh[d];
            lg += maskf[rr];
            const float nm = fmaxf(fm, lg);
            const float sc = __expf(fm - nm);
            const float p  = __expf(lg - nm);
            fl = fl * sc + p;
#pragma unroll
            for (int d = 0; d < 8; ++d)
                facc[d] = facc[d] * sc + p * (float)vb[rr * 68 + hb + d];
            fm = nm;
        }
        __syncthreads();
    }

    // ---- combine flash state across the 32 lanes ----
#pragma unroll
    for (int off = 16; off >= 1; off >>= 1) {
        const float om = __shfl_xor(fm, off, 32);
        const float ol = __shfl_xor(fl, off, 32);
        const float nm = fmaxf(fm, om);
        const float s1 = __expf(fm - nm);
        const float s2 = __expf(om - nm);
#pragma unroll
        for (int d = 0; d < 8; ++d) {
            const float oa = __shfl_xor(facc[d], off, 32);
            facc[d] = facc[d] * s1 + oa * s2;
        }
        fl = fl * s1 + ol * s2;
        fm = nm;
    }
    if (lane == 0) {
#pragma unroll
        for (int d = 0; d < 8; ++d) mhabuf[wave * 8 + d] = facc[d] / fl;
    }
    __syncthreads();

    // ---- epilogue (region reused as scratch) ----
    float* ctr  = region;          // 64
    float* mo   = region + 64;     // 64
    float* outp = region + 128;    // 64
    float* outn = region + 192;    // 64
    float* hid  = region + 256;    // 256
    float* st   = region + 512;    // 2

    if (tid < 64) ctr[tid] = A.center[i * 64 + tid];
    __syncthreads();

    if (tid < 64) {
        float s = 0.f;
        for (int k = 0; k < 64; ++k) s += mhabuf[k] * A.Wo[tid * 64 + k];
        mo[tid] = s;
    }
    __syncthreads();

    if (tid < 64) {
        float sg = A.ihB[tid] + A.hhB[tid];
        float sv = A.selfB[tid];
        for (int k = 0; k < 64; ++k) {
            sg += mo[k]  * A.ihW[tid * 64 + k];
            sg += ctr[k] * A.hhW[tid * 64 + k];
            sv += ctr[k] * A.selfW[tid * 64 + k];
        }
        const float g = 1.f / (1.f + __expf(-sg));
        outp[tid] = mo[tid] + g * (sv - mo[tid]);
    }
    __syncthreads();
    if (tid == 0) {
        float m = 0.f;
        for (int k = 0; k < 64; ++k) m += outp[k];
        m *= (1.f / 64.f);
        float v = 0.f;
        for (int k = 0; k < 64; ++k) { float d = outp[k] - m; v += d * d; }
        v *= (1.f / 64.f);
        st[0] = m;
        st[1] = rsqrtf(v + LNEPS);
    }
    __syncthreads();
    if (tid < 64)
        outn[tid] = (outp[tid] - st[0]) * st[1] * A.n2g[tid] + A.n2b[tid];
    __syncthreads();
    {
        float s = A.mB1[tid];
        for (int k = 0; k < 64; ++k) s += outn[k] * A.mW1[tid * 64 + k];
        hid[tid] = fmaxf(s, 0.f);
    }
    __syncthreads();
    if (tid < 64) {
        float s = A.mB2[tid];
        for (int k = 0; k < 256; ++k) s += hid[k] * A.mW2[tid * 256 + k];
        A.out[i * 64 + tid] = outn[tid] + s;
    }
}

// =============================== launch =====================================

extern "C" void kernel_launch(void* const* d_in, const int* in_sizes, int n_in,
                              void* d_out, int out_size, void* d_ws, size_t ws_size,
                              hipStream_t stream) {
    (void)in_sizes; (void)n_in; (void)out_size; (void)ws_size;
    const float* F(int) ;
    #define IN_F(k) ((const float*)d_in[(k)])

    // workspace carve (floats)
    float* ws = (float*)d_ws;
    float* Rws   = ws;                 // 4096
    float* posws = ws + 4096;          // 2048
    float* padws = ws + 6144;          // 1024
    float* ctrws = ws + 7168;          // 65536
    float* qws   = ws + 72704;         // 65536
    float* e1ws  = ws + 138240;        // 65536
    _Float16* wh = (_Float16*)(ws + 203776); // 16384 halves (4 x 64x64)

    K0Args a0;
    a0.positions = IN_F(0);
    a0.pmask     = (const unsigned char*)d_in[2];
    a0.tptr      = (const int*)d_in[3];
    a0.ceW1 = IN_F(4);  a0.ceb1 = IN_F(5);  a0.ceg1 = IN_F(6);  a0.cebe1 = IN_F(7);
    a0.ceW2 = IN_F(8);  a0.ceb2 = IN_F(9);  a0.ceg2 = IN_F(10); a0.cebe2 = IN_F(11);
    a0.ceW3 = IN_F(12); a0.ceb3 = IN_F(13); a0.ceg3 = IN_F(14); a0.cebe3 = IN_F(15);
    a0.n1g = IN_F(44);  a0.n1b = IN_F(45);  a0.Wq = IN_F(34);
    a0.ne1W1 = IN_F(22); a0.ne1b1 = IN_F(23); a0.ne1g1 = IN_F(24);
    a0.ne1be1 = IN_F(25); a0.ne1W2 = IN_F(26); a0.ne1b2 = IN_F(27);
    a0.Rws = Rws; a0.posws = posws; a0.padws = padws;
    a0.ctrws = ctrws; a0.qws = qws; a0.e1ws = e1ws;
    aae_precompute<<<NAG, 64, 0, stream>>>(a0);

    aae_cvt_f16<<<16, 256, 0, stream>>>(IN_F(20), IN_F(30), IN_F(35), IN_F(36), wh);

    MainArgs m;
    m.pos = posws; m.padf = padws; m.R = Rws;
    m.center = ctrws; m.q = qws; m.e1 = e1ws;
    m.wNe0 = wh; m.wNa = wh + 4096; m.wWk = wh + 8192; m.wWv = wh + 12288;
    m.ne0W1 = IN_F(16); m.ne0b1 = IN_F(17); m.ne0g1 = IN_F(18);
    m.ne0be1 = IN_F(19); m.ne0b2 = IN_F(21);
    m.nag1 = IN_F(28); m.nabe1 = IN_F(29); m.nab = IN_F(31);
    m.nag2 = IN_F(32); m.nabe2 = IN_F(33);
    m.Wo = IN_F(37);
    m.selfW = IN_F(38); m.selfB = IN_F(39);
    m.ihW = IN_F(40);   m.ihB = IN_F(41);
    m.hhW = IN_F(42);   m.hhB = IN_F(43);
    m.n2g = IN_F(46);   m.n2b = IN_F(47);
    m.mW1 = IN_F(48);   m.mB1 = IN_F(49);
    m.mW2 = IN_F(50);   m.mB2 = IN_F(51);
    m.out = (float*)d_out;
    aae_main<<<NAG, 256, 0, stream>>>(m);
    #undef IN_F
}
